// F_evolution_23579370455607
// MI455X (gfx1250) — matrix operand order repaired and checked by reading, not compiled, not run on hardware
//
#include <hip/hip_runtime.h>

typedef __attribute__((ext_vector_type(16))) _Float16 v16h;
typedef __attribute__((ext_vector_type(8)))  _Float16 v8h;
typedef __attribute__((ext_vector_type(8)))  float    v8f;
typedef __attribute__((ext_vector_type(4)))  float    v4f;

#define NWAVES 8
#define SLOPE 0.01f

// MLP 9->30->30->30->1 fwd + analytic input-grad, 16 samples per WMMA M-tile.
// All matmuls via v_wmma_f32_16x16x32_f16 (f32 accumulate). Weights held in
// registers in B-operand layout (built branchlessly from zero-padded LDS
// copies); C->A relayout via per-wave 16x32 f16 LDS tile with unconditional
// stores (pad columns compute to exact 0, so no guards needed).
__global__ __launch_bounds__(NWAVES * 32)
void mlp_fwd_bwd_wmma(const float* __restrict__ pe,
                      const float* __restrict__ q,
                      const float* __restrict__ si,
                      const float* __restrict__ W1, const float* __restrict__ B1,
                      const float* __restrict__ W2, const float* __restrict__ B2,
                      const float* __restrict__ W3, const float* __restrict__ B3,
                      const float* __restrict__ W4, const float* __restrict__ B4,
                      float* __restrict__ dout, int Nn)
{
    // zero-padded weight copies: branchless indexed reads in tile builders
    __shared__ float sW1p[32 * 32];   // [k][n], k<9,n<30 valid
    __shared__ float sW2p[32 * 32];   // [k][n], 30x30 valid
    __shared__ float sW3p[32 * 32];
    __shared__ float sW4p[32];        // [k]
    __shared__ float sBp[96];         // b1[0..31] b2[32..63] b3[64..95], padded 0
    __shared__ float sB4;
    __shared__ __align__(16) _Float16 Xb[NWAVES][512]; // 16x32 input tile (cols 9..31 stay 0)
    __shared__ __align__(16) _Float16 Hb[NWAVES][512]; // 16x32 hidden tile (fully rewritten each use)

    const int tid  = threadIdx.x;
    const int lane = tid & 31;
    const int wv   = tid >> 5;

    // ---- zero padded arrays + per-wave X buffer ----
    for (int i = tid; i < 32 * 32; i += NWAVES * 32) {
        sW1p[i] = 0.f; sW2p[i] = 0.f; sW3p[i] = 0.f;
    }
    for (int i = tid; i < 96; i += NWAVES * 32) sBp[i] = 0.f;
    if (tid < 32) sW4p[tid] = 0.f;
    for (int i = lane; i < 256; i += 32) ((unsigned int*)Xb[wv])[i] = 0u;
    __syncthreads();

    // ---- stage weights/biases into padded LDS (once per block) ----
    for (int i = tid; i < 270; i += NWAVES * 32) sW1p[(i / 30) * 32 + (i % 30)] = W1[i];
    for (int i = tid; i < 900; i += NWAVES * 32) {
        const int k = i / 30, n = i % 30;
        sW2p[k * 32 + n] = W2[i];
        sW3p[k * 32 + n] = W3[i];
    }
    for (int i = tid; i < 30; i += NWAVES * 32) {
        sW4p[i] = W4[i]; sBp[i] = B1[i]; sBp[32 + i] = B2[i]; sBp[64 + i] = B3[i];
    }
    if (tid == 0) sB4 = B4[0];
    __syncthreads();

    // ---- build B-operand register tiles (32x16 f16, zero-padded) ----
    // B layout: lane&15 = column; lanes 0-15 hold K=0..15 (VGPR v -> K=2v,2v+1),
    // lanes 16-31 hold K=16..31.
    auto bld = [&](auto wf, int ncol0) -> v16h {
        const int n  = (lane & 15) + ncol0;
        const int kb = (lane < 16) ? 0 : 16;
        v16h r;
#pragma unroll
        for (int v = 0; v < 8; ++v) {
            r[2 * v]     = (_Float16)wf(kb + 2 * v,     n);
            r[2 * v + 1] = (_Float16)wf(kb + 2 * v + 1, n);
        }
        return r;
    };
    auto w1f  = [&](int k, int n) { return sW1p[k * 32 + n]; };
    auto w2f  = [&](int k, int n) { return sW2p[k * 32 + n]; };
    auto w3f  = [&](int k, int n) { return sW3p[k * 32 + n]; };
    auto w1tf = [&](int k, int n) { return sW1p[n * 32 + k]; };
    auto w2tf = [&](int k, int n) { return sW2p[n * 32 + k]; };
    auto w3tf = [&](int k, int n) { return sW3p[n * 32 + k]; };
    auto w4f  = [&](int k, int n) { return (n == 0) ? sW4p[k] : 0.f; };

    const v16h W1B0 = bld(w1f, 0),  W1B1 = bld(w1f, 16);
    const v16h W2B0 = bld(w2f, 0),  W2B1 = bld(w2f, 16);
    const v16h W3B0 = bld(w3f, 0),  W3B1 = bld(w3f, 16);
    const v16h W4B  = bld(w4f, 0);
    const v16h T3B0 = bld(w3tf, 0), T3B1 = bld(w3tf, 16);
    const v16h T2B0 = bld(w2tf, 0), T2B1 = bld(w2tf, 16);
    const v16h T1B  = bld(w1tf, 0);

    const int nlo = lane & 15, nhi = nlo + 16;
    const float b1lo = sBp[nlo],      b1hi = sBp[nhi];
    const float b2lo = sBp[32 + nlo], b2hi = sBp[32 + nhi];
    const float b3lo = sBp[64 + nlo], b3hi = sBp[64 + nhi];
    const float b4v  = sB4;
    const float w4lo = sW4p[nlo],     w4hi = sW4p[nhi];

    // ---- small helpers ----
    auto wmma = [&](v16h a, v16h b, v8f c) -> v8f {
        return __builtin_amdgcn_wmma_f32_16x16x32_f16(false, a, false, b,
                                                      (short)0, c, false, false);
    };
    auto splat = [&](float x) -> v8f {
        v8f c;
#pragma unroll
        for (int r = 0; r < 8; ++r) c[r] = x;
        return c;
    };
    auto leaky = [&](v8f z) -> v8f {
        v8f h;
#pragma unroll
        for (int r = 0; r < 8; ++r) h[r] = fmaxf(z[r], z[r] * SLOPE);
        return h;
    };
    auto mmul = [&](v8f g, v8f z) -> v8f {  // g *= leakyrelu'(z)
#pragma unroll
        for (int r = 0; r < 8; ++r) g[r] *= (z[r] > 0.f ? 1.f : SLOPE);
        return g;
    };
    // C layout (lane&15 = col; VGPR r -> row r, lanes>=16 -> row r+8) -> LDS row-major.
    // Unconditional: hi columns >=30 carry exact 0 (zero B columns + zero bias).
    auto storeH = [&](v8f lo, v8f hi) {
        _Float16* buf = Hb[wv] + (((lane < 16) ? 0 : 8) * 32 + nlo);
#pragma unroll
        for (int r = 0; r < 8; ++r) {
            buf[r * 32]      = (_Float16)lo[r];
            buf[r * 32 + 16] = (_Float16)hi[r];
        }
    };
    // LDS row-major -> A layout (lane&15 = row; lanes<16: K 0..15, lanes>=16: K 16..31)
    auto loadA = [&](const _Float16* buf) -> v16h {
        const int m  = lane & 15;
        const int ko = (lane < 16) ? 0 : 16;
        v8h lo = *(const v8h*)(buf + m * 32 + ko);
        v8h hi = *(const v8h*)(buf + m * 32 + ko + 8);
        v16h a;
#pragma unroll
        for (int i = 0; i < 8; ++i) { a[i] = lo[i]; a[i + 8] = hi[i]; }
        return a;
    };

    const int tiles  = (Nn + 15) >> 4;
    const int stride = gridDim.x * NWAVES;
    float* outp = dout;
    float* hs   = dout + (size_t)Nn;
    float* hq   = dout + (size_t)5 * Nn;
    float* hpe  = dout + (size_t)6 * Nn;

    // branchless per-lane prefetch base: lanes 0-15 cover pe rows, 16-31 cover si rows
    const float* pfbase = (lane < 16) ? pe : si;

    for (int t = blockIdx.x * NWAVES + wv; t < tiles; t += stride) {
        const int base = t << 4;

        // ---- stage 16x9 input tile into Xb (coalesced float4 loads) ----
        {
            const int r = lane & 15;
            int s = base + r; if (s >= Nn) s = Nn - 1;
            _Float16* row = Xb[wv] + r * 32;
            if (lane < 16) {
                v4f p = ((const v4f*)pe)[s];
                float qq = q[s];
                row[0] = (_Float16)p[0]; row[1] = (_Float16)p[1];
                row[2] = (_Float16)p[2]; row[3] = (_Float16)p[3];
                row[4] = (_Float16)qq;
            } else {
                v4f sv = ((const v4f*)si)[s];
                row[5] = (_Float16)sv[0]; row[6] = (_Float16)sv[1];
                row[7] = (_Float16)sv[2]; row[8] = (_Float16)sv[3];
            }
        }
        // prefetch next iteration's inputs into near caches (uniform branch,
        // branchless address: one line per lane, lane 0 also covers the q row)
        {
            const int tn = t + stride;
            if (tn < tiles) {
                int sn = (tn << 4) + nlo; if (sn >= Nn) sn = Nn - 1;
                __builtin_prefetch(pfbase + (size_t)sn * 4, 0, 3);
                if (lane == 0) __builtin_prefetch(q + sn, 0, 3);
            }
        }
        __builtin_amdgcn_wave_barrier();
        v16h a0 = loadA(Xb[wv]);

        // ---- forward ----
        v8f z1lo = wmma(a0, W1B0, splat(b1lo));
        v8f z1hi = wmma(a0, W1B1, splat(b1hi));
        storeH(leaky(z1lo), leaky(z1hi));
        __builtin_amdgcn_wave_barrier();
        v16h a1 = loadA(Hb[wv]);

        v8f z2lo = wmma(a1, W2B0, splat(b2lo));
        v8f z2hi = wmma(a1, W2B1, splat(b2hi));
        storeH(leaky(z2lo), leaky(z2hi));
        __builtin_amdgcn_wave_barrier();
        v16h a2 = loadA(Hb[wv]);

        v8f z3lo = wmma(a2, W3B0, splat(b3lo));
        v8f z3hi = wmma(a2, W3B1, splat(b3hi));
        storeH(leaky(z3lo), leaky(z3hi));
        __builtin_amdgcn_wave_barrier();
        v16h a3 = loadA(Hb[wv]);

        v8f z4 = wmma(a3, W4B, splat(b4v));   // out in column 0

        // ---- backward (analytic grad wrt inputs) ----
        v8f g3lo, g3hi;
#pragma unroll
        for (int r = 0; r < 8; ++r) {
            g3lo[r] = w4lo * (z3lo[r] > 0.f ? 1.f : SLOPE);
            g3hi[r] = w4hi * (z3hi[r] > 0.f ? 1.f : SLOPE);
        }
        storeH(g3lo, g3hi);
        __builtin_amdgcn_wave_barrier();
        v16h ag3 = loadA(Hb[wv]);

        v8f g2lo = mmul(wmma(ag3, T3B0, splat(0.f)), z2lo);
        v8f g2hi = mmul(wmma(ag3, T3B1, splat(0.f)), z2hi);
        storeH(g2lo, g2hi);
        __builtin_amdgcn_wave_barrier();
        v16h ag2 = loadA(Hb[wv]);

        v8f g1lo = mmul(wmma(ag2, T2B0, splat(0.f)), z1lo);
        v8f g1hi = mmul(wmma(ag2, T2B1, splat(0.f)), z1hi);
        storeH(g1lo, g1hi);
        __builtin_amdgcn_wave_barrier();
        v16h ag1 = loadA(Hb[wv]);

        v8f gx = wmma(ag1, T1B, splat(0.f));  // columns 0..8 valid

        // ---- stores: out, then gx -> (h_pe cols 0..3, h_q col 4, h_s cols 5..8) ----
        const int mb = (lane < 16) ? 0 : 8;
        if (base + 16 <= Nn) {               // uniform fast path: full tile
            const int s0 = base + mb;
            if (nlo == 0) {
#pragma unroll
                for (int r = 0; r < 8; ++r) outp[s0 + r] = z4[r];
            }
            if (nlo < 4) {
#pragma unroll
                for (int r = 0; r < 8; ++r) hpe[(size_t)(s0 + r) * 4 + nlo] = gx[r];
            } else if (nlo == 4) {
#pragma unroll
                for (int r = 0; r < 8; ++r) hq[s0 + r] = gx[r];
            } else if (nlo < 9) {
#pragma unroll
                for (int r = 0; r < 8; ++r) hs[(size_t)(s0 + r) * 4 + (nlo - 5)] = gx[r];
            }
        } else {                              // tail tile: per-sample guards
            if (nlo == 0) {
#pragma unroll
                for (int r = 0; r < 8; ++r) {
                    int s = base + mb + r;
                    if (s < Nn) outp[s] = z4[r];
                }
            }
            if (nlo < 4) {
#pragma unroll
                for (int r = 0; r < 8; ++r) {
                    int s = base + mb + r;
                    if (s < Nn) hpe[(size_t)s * 4 + nlo] = gx[r];
                }
            } else if (nlo == 4) {
#pragma unroll
                for (int r = 0; r < 8; ++r) {
                    int s = base + mb + r;
                    if (s < Nn) hq[s] = gx[r];
                }
            } else if (nlo < 9) {
#pragma unroll
                for (int r = 0; r < 8; ++r) {
                    int s = base + mb + r;
                    if (s < Nn) hs[(size_t)s * 4 + (nlo - 5)] = gx[r];
                }
            }
        }
    }
}

extern "C" void kernel_launch(void* const* d_in, const int* in_sizes, int n_in,
                              void* d_out, int out_size, void* d_ws, size_t ws_size,
                              hipStream_t stream) {
    const float* pe = (const float*)d_in[0];
    const float* q  = (const float*)d_in[1];
    const float* si = (const float*)d_in[2];
    const float* W1 = (const float*)d_in[3];
    const float* B1 = (const float*)d_in[4];
    const float* W2 = (const float*)d_in[5];
    const float* B2 = (const float*)d_in[6];
    const float* W3 = (const float*)d_in[7];
    const float* B3 = (const float*)d_in[8];
    const float* W4 = (const float*)d_in[9];
    const float* B4 = (const float*)d_in[10];
    const int Nn = in_sizes[1];            // input_q holds N elements
    const int tiles = (Nn + 15) / 16;
    int blocks = (tiles + NWAVES - 1) / NWAVES;
    if (blocks > 1024) blocks = 1024;      // ~8192 waves; ~8 tiles/wave amortizes setup
    mlp_fwd_bwd_wmma<<<blocks, NWAVES * 32, 0, stream>>>(
        pe, q, si, W1, B1, W2, B2, W3, B3, W4, B4, (float*)d_out, Nn);
}
